// HyperClassifier_65292092833793
// MI455X (gfx1250) — matrix operand order repaired and checked by reading, not compiled, not run on hardware
//
#include <hip/hip_runtime.h>
#include <hip/hip_bf16.h>

typedef __attribute__((ext_vector_type(16))) _Float16 v16h;
typedef __attribute__((ext_vector_type(8)))  _Float16 v8h;
typedef __attribute__((ext_vector_type(8)))  float    v8f;

#define WAVES 8
#define TPB   (WAVES * 32)

#define MODE_EDGE 0
#define MODE_N1   1
#define MODE_NODE 2

static constexpr int Nn = 50000;
static constexpr int Ee = 1600000;

// ---------------------------------------------------------------------------
// Pack f32 row-major W[kdim][64] into the CDNA5 WMMA B-fragment lane layout.
// Packed index p = ((c*4+nc)*32 + lane)*16 + j  maps to
//   k = c*32 + (lane>>4)*8 + (j&7) + (j>>3)*16 ,  n = nc*16 + (lane&15)
// so each lane's 16-f16 B fragment is one contiguous 32-byte run.
// ---------------------------------------------------------------------------
__global__ void pack_w(const float* __restrict__ src, _Float16* __restrict__ dst, int kdim) {
    int p = blockIdx.x * blockDim.x + threadIdx.x;
    if (p >= kdim * 64) return;
    int j    = p & 15;
    int lane = (p >> 4) & 31;
    int nc   = (p >> 9) & 3;
    int c    = p >> 11;
    int k = c * 32 + (lane >> 4) * 8 + (j & 7) + (j >> 3) * 16;
    int n = nc * 16 + (lane & 15);
    dst[p] = (_Float16)src[k * 64 + n];
}

__global__ void zero_f32(float* __restrict__ p, int n) {
    int i = blockIdx.x * blockDim.x + threadIdx.x;
    if (i < n) p[i] = 0.0f;
}

__global__ void count_deg(const int* __restrict__ dst, float* __restrict__ cnt, int e) {
    int i = blockIdx.x * blockDim.x + threadIdx.x;
    if (i < e) atomicAdd(&cnt[dst[i]], 1.0f);
}

// ---------------------------------------------------------------------------
// Fused per-row MLP:  Z(16 x din) -> GEMM1 -> +b1 -> LeakyReLU -> LayerNorm
//                       -> GEMM2 -> +b2 -> store (EDGE/NODE) / atomic (N1).
// One 16-row tile per wave; both GEMMs via v_wmma_f32_16x16x32_f16.
// MODE/NF/EW are compile-time: each 32-wide k-chunk sits in one source
// segment, so the gather source resolves at compile time after unrolling.
// ---------------------------------------------------------------------------
template <int MODE, int NF, int EW>
__global__ __launch_bounds__(TPB) void fused_mlp(
    int rows,
    const float* __restrict__ xin,
    const float* __restrict__ ein,
    const int* __restrict__ srcI, const int* __restrict__ dstI,
    const float* __restrict__ cnt,
    const _Float16* __restrict__ W1pk, const _Float16* __restrict__ W2pk,
    const float* __restrict__ b1, const float* __restrict__ gam,
    const float* __restrict__ bet, const float* __restrict__ b2,
    float* __restrict__ outp)
{
    constexpr int DIN = (MODE == MODE_EDGE) ? (2 * NF + EW) : (NF + EW);
    constexpr int NK1 = DIN / 32;
    constexpr bool ATOMIC = (MODE == MODE_N1);

    __shared__ __align__(32) _Float16 W1s[DIN * 64];
    __shared__ __align__(32) _Float16 W2s[64 * 64];
    __shared__ __align__(32) _Float16 Hs[WAVES][16 * 64];
    __shared__ float b1s[64], gs[64], bes[64], b2s[64];

    const int tid  = threadIdx.x;
    const int wave = tid >> 5;
    const int lane = tid & 31;
    const int nl   = lane & 15;
    const int khi  = lane >> 4;

    // ---- stage packed weights + vectors into LDS (uint4 = 8 f16) ----
    {
        const uint4* s1 = (const uint4*)W1pk;
        uint4*       d1 = (uint4*)W1s;
#pragma unroll 4
        for (int i = tid; i < DIN * 8; i += TPB) d1[i] = s1[i];
        const uint4* s2 = (const uint4*)W2pk;
        uint4*       d2 = (uint4*)W2s;
#pragma unroll 2
        for (int i = tid; i < 512; i += TPB) d2[i] = s2[i];
        if (tid < 64) { b1s[tid] = b1[tid]; gs[tid] = gam[tid]; bes[tid] = bet[tid]; b2s[tid] = b2[tid]; }
    }
    __syncthreads();

    const int tiles = rows >> 4;
    int tile = blockIdx.x * WAVES + wave;
    const bool valid = (tile < tiles);
    if (!valid) tile = 0;

    const int rowm = tile * 16 + nl;     // this lane's A-matrix row
    int sidx = 0, didx = 0;
    float inv = 1.0f;
    if (MODE != MODE_NODE) { sidx = srcI[rowm]; didx = dstI[rowm]; }
    else                   { inv = 1.0f / fmaxf(cnt[rowm], 1.0f); }

    // per-lane row base pointers (branch-free inner loop)
    const float* pxs = (MODE == MODE_NODE) ? xin + (size_t)rowm * NF
                                           : xin + (size_t)sidx * NF;
    const float* pxd = (MODE == MODE_EDGE) ? xin + (size_t)didx * NF : nullptr;
    const float* pe  = ein + (size_t)rowm * EW;

    // ---- GEMM1: acc[nc] (16x16 f32) over DIN/32 k-chunks ----
    const v8f vzero = {0.f, 0.f, 0.f, 0.f, 0.f, 0.f, 0.f, 0.f};
    v8f acc[4] = {vzero, vzero, vzero, vzero};
#pragma unroll
    for (int c = 0; c < NK1; ++c) {
        const float* p;
        float sc = 1.0f;
        if (MODE == MODE_EDGE) {
            if (c * 32 < NF)            p = pxs + c * 32;
            else if (c * 32 < 2 * NF)   p = pxd + c * 32 - NF;
            else                        p = pe  + c * 32 - 2 * NF;
        } else {
            if (c * 32 < NF)            p = pxs + c * 32;
            else { p = pe + c * 32 - NF; if (MODE == MODE_NODE) sc = inv; }
        }
        p += khi * 8;
        const float4 a0 = *(const float4*)(p);
        const float4 a1 = *(const float4*)(p + 4);
        const float4 c0 = *(const float4*)(p + 16);
        const float4 c1 = *(const float4*)(p + 20);
        v16h a;
        a[0]  = (_Float16)(a0.x * sc); a[1]  = (_Float16)(a0.y * sc);
        a[2]  = (_Float16)(a0.z * sc); a[3]  = (_Float16)(a0.w * sc);
        a[4]  = (_Float16)(a1.x * sc); a[5]  = (_Float16)(a1.y * sc);
        a[6]  = (_Float16)(a1.z * sc); a[7]  = (_Float16)(a1.w * sc);
        a[8]  = (_Float16)(c0.x * sc); a[9]  = (_Float16)(c0.y * sc);
        a[10] = (_Float16)(c0.z * sc); a[11] = (_Float16)(c0.w * sc);
        a[12] = (_Float16)(c1.x * sc); a[13] = (_Float16)(c1.y * sc);
        a[14] = (_Float16)(c1.z * sc); a[15] = (_Float16)(c1.w * sc);
#pragma unroll
        for (int nc = 0; nc < 4; ++nc) {
            const v16h b = *(const v16h*)&W1s[((c * 4 + nc) * 32 + lane) * 16];
            acc[nc] = __builtin_amdgcn_wmma_f32_16x16x32_f16(
                false, a, false, b, (short)0, acc[nc], false, false);
        }
    }

    // ---- epilogue: +b1, LeakyReLU, LayerNorm (per-row stats via shfl) ----
    // C/D layout: lane holds rows M = khi*8 + r (r=0..7), column n = nc*16 + nl.
    float hv[4][8];
#pragma unroll
    for (int nc = 0; nc < 4; ++nc)
#pragma unroll
        for (int r = 0; r < 8; ++r) {
            float h = acc[nc][r] + b1s[nc * 16 + nl];
            hv[nc][r] = (h > 0.0f) ? h : 0.01f * h;
        }
    float s[8], q[8];
#pragma unroll
    for (int r = 0; r < 8; ++r) {
        s[r] = 0.0f; q[r] = 0.0f;
#pragma unroll
        for (int nc = 0; nc < 4; ++nc) { s[r] += hv[nc][r]; q[r] += hv[nc][r] * hv[nc][r]; }
    }
#pragma unroll
    for (int m = 1; m < 16; m <<= 1)
#pragma unroll
        for (int r = 0; r < 8; ++r) {
            s[r] += __shfl_xor(s[r], m, 32);
            q[r] += __shfl_xor(q[r], m, 32);
        }
    float mu[8], rstd[8];
#pragma unroll
    for (int r = 0; r < 8; ++r) {
        mu[r] = s[r] * (1.0f / 64.0f);
        float var = q[r] * (1.0f / 64.0f) - mu[r] * mu[r];
        rstd[r] = rsqrtf(var + 1e-5f);
    }
#pragma unroll
    for (int nc = 0; nc < 4; ++nc) {
        const float gg = gs[nc * 16 + nl], bb = bes[nc * 16 + nl];
#pragma unroll
        for (int r = 0; r < 8; ++r) {
            float h = (hv[nc][r] - mu[r]) * rstd[r] * gg + bb;
            Hs[wave][(khi * 8 + r) * 64 + nc * 16 + nl] = (_Float16)h;
        }
    }
    __syncthreads();   // make H visible for re-fragmenting as A of GEMM2

    // ---- GEMM2: (16x64) x (64x64), K = 64 -> 2 k-chunks ----
    v8f oacc[4] = {vzero, vzero, vzero, vzero};
#pragma unroll
    for (int c = 0; c < 2; ++c) {
        const _Float16* base = &Hs[wave][nl * 64 + c * 32 + khi * 8];
        v8h lo = *(const v8h*)base;
        v8h hi = *(const v8h*)(base + 16);
        v16h a;
#pragma unroll
        for (int i = 0; i < 8; ++i) { a[i] = lo[i]; a[8 + i] = hi[i]; }
#pragma unroll
        for (int nc = 0; nc < 4; ++nc) {
            const v16h b = *(const v16h*)&W2s[((c * 4 + nc) * 32 + lane) * 16];
            oacc[nc] = __builtin_amdgcn_wmma_f32_16x16x32_f16(
                false, a, false, b, (short)0, oacc[nc], false, false);
        }
    }

    // ---- +b2, then store rows (EDGE/NODE) or atomic scatter-add (N1) ----
    if (valid) {
#pragma unroll
        for (int nc = 0; nc < 4; ++nc) {
            const int col = nc * 16 + nl;
            const float bb = b2s[col];
#pragma unroll
            for (int r = 0; r < 8; ++r) {
                const float val = oacc[nc][r] + bb;
                const int rloc = khi * 8 + r;
                if (ATOMIC) {
                    const int d = __shfl(didx, rloc, 32);   // dst index of that row
                    atomicAdd(&outp[(size_t)d * 64 + col], val);
                } else {
                    outp[(size_t)(tile * 16 + rloc) * 64 + col] = val;
                }
            }
        }
    }
}

// ---------------------------------------------------------------------------
extern "C" void kernel_launch(void* const* d_in, const int* in_sizes, int n_in,
                              void* d_out, int out_size, void* d_ws, size_t ws_size,
                              hipStream_t stream) {
    const float* x0  = (const float*)d_in[0];
    const float* ea0 = (const float*)d_in[1];
    const int* eidx  = (const int*)d_in[2];
    const int* srcI  = eidx;
    const int* dstI  = eidx + Ee;

    // params flatten: for l in 0..2, for m in {edge,n1,n2}, tensors W1,b1,g,be,W2,b2
    const float* P[3][3][6];
    int idx = 3;
    for (int l = 0; l < 3; ++l)
        for (int m = 0; m < 3; ++m)
            for (int t = 0; t < 6; ++t)
                P[l][m][t] = (const float*)d_in[idx++];

    const int dinW1[3][3] = {{96, 96, 96}, {192, 128, 128}, {192, 128, 128}};

    // workspace carve (all offsets 16B aligned)
    char* wp = (char*)d_ws;
    float* agg = (float*)wp; wp += (size_t)Nn * 64 * sizeof(float);
    float* cnt = (float*)wp; wp += (size_t)Nn * sizeof(float);
    float* xws = (float*)wp; wp += (size_t)Nn * 64 * sizeof(float);
    _Float16* pk1[3][3];
    _Float16* pk2[3][3];
    for (int l = 0; l < 3; ++l)
        for (int m = 0; m < 3; ++m) {
            pk1[l][m] = (_Float16*)wp; wp += (size_t)dinW1[l][m] * 64 * sizeof(_Float16);
            pk2[l][m] = (_Float16*)wp; wp += (size_t)64 * 64 * sizeof(_Float16);
        }

    // pack weights into WMMA B-fragment layout (f32 -> f16)
    for (int l = 0; l < 3; ++l)
        for (int m = 0; m < 3; ++m) {
            const int kd = dinW1[l][m];
            pack_w<<<(kd * 64 + 255) / 256, 256, 0, stream>>>(P[l][m][0], pk1[l][m], kd);
            pack_w<<<(64 * 64 + 255) / 256, 256, 0, stream>>>(P[l][m][4], pk2[l][m], 64);
        }

    // in-degree counts (fixed across layers)
    zero_f32<<<(Nn + 255) / 256, 256, 0, stream>>>(cnt, Nn);
    count_deg<<<(Ee + 255) / 256, 256, 0, stream>>>(dstI, cnt, Ee);

    float* outX  = (float*)d_out;                   // N x 64
    float* eabuf = (float*)d_out + (size_t)Nn * 64; // E x 64 (final edge_attr, reused in-place)

    const int tilesE = Ee / 16, tilesN = Nn / 16;
    const int gE = (tilesE + WAVES - 1) / WAVES;
    const int gN = (tilesN + WAVES - 1) / WAVES;

    const float* xcur = x0;

    for (int l = 0; l < 3; ++l) {
        const float* eacur = (l == 0) ? ea0 : eabuf;

        // EdgeModel: cat(x[src], x[dst], ea) -> MLP -> ea (in-place after layer 0)
        if (l == 0)
            fused_mlp<MODE_EDGE, 32, 32><<<gE, TPB, 0, stream>>>(
                Ee, xcur, eacur, srcI, dstI, nullptr,
                pk1[l][0], pk2[l][0], P[l][0][1], P[l][0][2], P[l][0][3], P[l][0][5], eabuf);
        else
            fused_mlp<MODE_EDGE, 64, 64><<<gE, TPB, 0, stream>>>(
                Ee, xcur, eacur, srcI, dstI, nullptr,
                pk1[l][0], pk2[l][0], P[l][0][1], P[l][0][2], P[l][0][3], P[l][0][5], eabuf);

        // scatter-mean accumulator
        zero_f32<<<(Nn * 64 + 255) / 256, 256, 0, stream>>>(agg, Nn * 64);

        // NodeModel stage 1: cat(x[src], ea_new) -> MLP -> atomic scatter-add by dst
        if (l == 0)
            fused_mlp<MODE_N1, 32, 64><<<gE, TPB, 0, stream>>>(
                Ee, xcur, eabuf, srcI, dstI, nullptr,
                pk1[l][1], pk2[l][1], P[l][1][1], P[l][1][2], P[l][1][3], P[l][1][5], agg);
        else
            fused_mlp<MODE_N1, 64, 64><<<gE, TPB, 0, stream>>>(
                Ee, xcur, eabuf, srcI, dstI, nullptr,
                pk1[l][1], pk2[l][1], P[l][1][1], P[l][1][2], P[l][1][3], P[l][1][5], agg);

        // NodeModel stage 2: cat(x, agg/deg) -> MLP -> x
        float* xout = (l == 2) ? outX : xws;
        if (l == 0)
            fused_mlp<MODE_NODE, 32, 64><<<gN, TPB, 0, stream>>>(
                Nn, xcur, agg, nullptr, nullptr, cnt,
                pk1[l][2], pk2[l][2], P[l][2][1], P[l][2][2], P[l][2][3], P[l][2][5], xout);
        else
            fused_mlp<MODE_NODE, 64, 64><<<gN, TPB, 0, stream>>>(
                Nn, xcur, agg, nullptr, nullptr, cnt,
                pk1[l][2], pk2[l][2], P[l][2][1], P[l][2][2], P[l][2][3], P[l][2][5], xout);

        xcur = xout;
    }
}
